// Mamba2Fusion_69355131895939
// MI455X (gfx1250) — matrix operand order repaired and checked by reading, not compiled, not run on hardware
//
#include <hip/hip_runtime.h>
#include <hip/hip_bf16.h>
#include <math.h>

typedef _Float16 h16;
typedef _Float16 v8h  __attribute__((ext_vector_type(8)));
typedef _Float16 v16h __attribute__((ext_vector_type(16)));
typedef float    v8f  __attribute__((ext_vector_type(8)));

#define DEV __device__ __forceinline__

// ---------------- workspace layout (halves) ----------------
// All weights stored transposed+f16: Wt[k*OUT + n] = (h16)W[n*K + k], K2 rows (>=K, zero pad).
static constexpr size_t OFF_BP   = 0;                       // 128x256
static constexpr size_t OFF_VP   = 32768;                   // 256x256
static constexpr size_t OFF_ATTN = 98304;                   // 8 x 256x256 (d*4 + {q,k,v,o})
static constexpr size_t OFF_FIN  = 622592;                  // 256x256
static constexpr size_t OFF_BLK  = 688128;
static constexpr size_t BLK_STRIDE = 733184;
static constexpr size_t B_IN = 0, B_GATE = 65536, B_SOUT = 131072,
                        B_F1 = 196608, B_F2 = 458752, B_BM = 720896, B_CT = 724992;

// ---------------- weight transpose+convert ----------------
__global__ void convT(const float* __restrict__ src, h16* __restrict__ dst,
                      int OUT, int K, int K2) {
  int idx = blockIdx.x * blockDim.x + threadIdx.x;
  if (idx >= K2 * OUT) return;
  int k = idx / OUT, n = idx - k * OUT;
  dst[idx] = (k < K) ? (h16)src[(size_t)n * K + k] : (h16)0.f;
}

#define SHUF16(lo, hi) __builtin_shufflevector((lo), (hi), 0,1,2,3,4,5,6,7,8,9,10,11,12,13,14,15)

// ---------------- dual-M WMMA: one B-fragment fetch feeds both 16-row tiles ----------------
// C0(16x16) = A[0:16,:] * B ; C1(16x16) = A[16:32,:] * B, f32 accum, K multiple of 32.
DEV void wmma_mm2(const h16* __restrict__ A, int lda,
                  const h16* __restrict__ B, int ldb,
                  int K, int n0, v8f& acc0, v8f& acc1) {
  const int lane = threadIdx.x & 31;
  const int lh = lane >> 4, ll = lane & 15;
  acc0 = {}; acc1 = {};
  const h16* ap0 = A + (size_t)ll * lda + (lh << 3);
  const h16* ap1 = ap0 + (size_t)16 * lda;
  const h16* bp0 = B + n0 + ll + (size_t)(lh << 3) * ldb;
  __builtin_prefetch(bp0, 0, 1);
  for (int k0 = 0; k0 < K; k0 += 32) {
    v16h b;
#pragma unroll
    for (int s = 0; s < 8; ++s) {
      b[s]     = bp0[(size_t)(k0 + s) * ldb];
      b[s + 8] = bp0[(size_t)(k0 + 16 + s) * ldb];
    }
    v8h a0lo = *(const v8h*)(ap0 + k0);
    v8h a0hi = *(const v8h*)(ap0 + k0 + 16);
    v16h a0 = SHUF16(a0lo, a0hi);
    acc0 = __builtin_amdgcn_wmma_f32_16x16x32_f16(false, a0, false, b,
                                                  (short)0, acc0, false, false);
    v8h a1lo = *(const v8h*)(ap1 + k0);
    v8h a1hi = *(const v8h*)(ap1 + k0 + 16);
    v16h a1 = SHUF16(a1lo, a1hi);
    acc1 = __builtin_amdgcn_wmma_f32_16x16x32_f16(false, a1, false, b,
                                                  (short)0, acc1, false, false);
  }
}

// ---------------- params ----------------
struct Params {
  const float *ble, *vis;
  const float *bp_b, *bp_emb, *bp_g, *bp_beta;
  const float *vp_b, *vp_emb, *vp_g, *vp_beta;
  const float *a_qb, *a_kb, *a_vb, *a_ob, *a_g, *a_beta;
  const float *m_ln1_g, *m_ln1_b, *m_in_b, *m_A, *m_gate_b, *m_sout_b;
  const float *m_ln2_g, *m_ln2_b, *m_f1_b, *m_f2_b;
  const float *fin_g, *fin_beta, *fin_w, *fin_b;
};

DEV float gelu_exact(float x) { return 0.5f * x * (1.f + erff(x * 0.70710678118f)); }

// LayerNorm over 32 rows of 256 (one row per wave per pass).
DEV void ln_rows(const float* __restrict__ X, const h16* __restrict__ addh,
                 const float* __restrict__ g, const float* __restrict__ b,
                 h16* __restrict__ dsth, float* __restrict__ dstf, bool accum) {
  const int wv = threadIdx.x >> 5, lane = threadIdx.x & 31;
  for (int t = wv; t < 32; t += 8) {
    const float* x = X + t * 256;
    float vals[8]; float s = 0.f;
#pragma unroll
    for (int q = 0; q < 8; ++q) {
      int j = lane + (q << 5);
      float v = x[j];
      if (addh) v += (float)addh[t * 256 + j];
      vals[q] = v; s += v;
    }
#pragma unroll
    for (int o = 16; o; o >>= 1) s += __shfl_xor(s, o);
    float mu = s * (1.f / 256.f);
    float vv = 0.f;
#pragma unroll
    for (int q = 0; q < 8; ++q) { float d = vals[q] - mu; vv += d * d; }
#pragma unroll
    for (int o = 16; o; o >>= 1) vv += __shfl_xor(vv, o);
    float rs = rsqrtf(vv * (1.f / 256.f) + 1e-5f);
#pragma unroll
    for (int q = 0; q < 8; ++q) {
      int j = lane + (q << 5);
      float y = (vals[q] - mu) * rs * g[j] + b[j];
      if (dsth) dsth[t * 256 + j] = (h16)y;
      if (dstf) { if (accum) dstf[t * 256 + j] += y; else dstf[t * 256 + j] = y; }
    }
  }
}

// ---------------- megakernel: one workgroup per batch element ----------------
__global__ __launch_bounds__(256, 2)
void fused_fwd(Params p, const h16* __restrict__ ws, float* __restrict__ out) {
  __shared__ __align__(16) h16  AH[32768];   // 64KB arena (Q|KT|V|w, Hs, FFN hidden)
  __shared__ __align__(16) h16  BH[16384];   // 32KB (ble_h/vis_h, xln/x_in, ys)
  __shared__ __align__(16) float RES[8192];  // 32KB residual stream (32x256)
  __shared__ __align__(16) float FB[8192];   // 32KB f32 scratch (gemm out / scores / gate / xB)
  __shared__ float hbuf[16];

  const int bid = blockIdx.x, tid = threadIdx.x;
  const int wv = tid >> 5, lane = tid & 31, lh = lane >> 4, ll = lane & 15;

  // ---- stage raw inputs as f16 into AH ----
  {
    const float* sb = p.ble + (size_t)bid * 32 * 128;
    for (int j = tid; j < 4096; j += 256) AH[j] = (h16)sb[j];
    const float* sv = p.vis + (size_t)bid * 32 * 256;
    for (int j = tid; j < 8192; j += 256) AH[4096 + j] = (h16)sv[j];
  }
  __syncthreads();

  // ---- ble projection: (32x128)@Wt -> FB (+bias+emb), LN -> BH[0:8192) ----
  for (int nt = wv; nt < 16; nt += 8) {
    int n0 = nt << 4, c = n0 + ll;
    v8f a0, a1; wmma_mm2(AH, 128, ws + OFF_BP, 256, 128, n0, a0, a1);
    float bb = p.bp_b[c] + p.bp_emb[c];
#pragma unroll
    for (int i = 0; i < 8; ++i) {
      FB[(i + (lh << 3)) * 256 + c]      = a0[i] + bb;
      FB[(16 + i + (lh << 3)) * 256 + c] = a1[i] + bb;
    }
  }
  __syncthreads();
  ln_rows(FB, nullptr, p.bp_g, p.bp_beta, BH, nullptr, false);
  __syncthreads();

  // ---- vis projection -> BH[8192:16384) ----
  for (int nt = wv; nt < 16; nt += 8) {
    int n0 = nt << 4, c = n0 + ll;
    v8f a0, a1; wmma_mm2(AH + 4096, 256, ws + OFF_VP, 256, 256, n0, a0, a1);
    float bb = p.vp_b[c] + p.vp_emb[c];
#pragma unroll
    for (int i = 0; i < 8; ++i) {
      FB[(i + (lh << 3)) * 256 + c]      = a0[i] + bb;
      FB[(16 + i + (lh << 3)) * 256 + c] = a1[i] + bb;
    }
  }
  __syncthreads();
  ln_rows(FB, nullptr, p.vp_g, p.vp_beta, BH + 8192, nullptr, false);
  __syncthreads();

  // ---- cross attention, two directions; fused = LN(dir0) + LN(dir1) into RES ----
  for (int d = 0; d < 2; ++d) {
    const h16* qh  = d ? BH + 8192 : BH;
    const h16* kvh = d ? BH : BH + 8192;
    const h16* wq  = ws + OFF_ATTN + (size_t)(d * 4 + 0) * 65536;
    const h16* wk  = ws + OFF_ATTN + (size_t)(d * 4 + 1) * 65536;
    const h16* wvp = ws + OFF_ATTN + (size_t)(d * 4 + 2) * 65536;
    const h16* wo  = ws + OFF_ATTN + (size_t)(d * 4 + 3) * 65536;
    const float *qb = p.a_qb + d * 256, *kb = p.a_kb + d * 256,
                *vb = p.a_vb + d * 256, *ob = p.a_ob + d * 256;

    // Q -> AH[0:8192), K^T -> AH[8192:16384) (channel-major), V -> AH[16384:24576)
    for (int nt = wv; nt < 16; nt += 8) {
      int n0 = nt << 4, c = n0 + ll;
      v8f a0, a1; wmma_mm2(qh, 256, wq, 256, 256, n0, a0, a1);
      float bb = qb[c];
#pragma unroll
      for (int i = 0; i < 8; ++i) {
        AH[(i + (lh << 3)) * 256 + c]      = (h16)(a0[i] + bb);
        AH[(16 + i + (lh << 3)) * 256 + c] = (h16)(a1[i] + bb);
      }
    }
    for (int nt = wv; nt < 16; nt += 8) {
      int n0 = nt << 4, c = n0 + ll;
      v8f a0, a1; wmma_mm2(kvh, 256, wk, 256, 256, n0, a0, a1);
      float bb = kb[c];
#pragma unroll
      for (int i = 0; i < 8; ++i) {
        AH[8192 + c * 32 + (i + (lh << 3))]      = (h16)(a0[i] + bb);
        AH[8192 + c * 32 + (16 + i + (lh << 3))] = (h16)(a1[i] + bb);
      }
    }
    for (int nt = wv; nt < 16; nt += 8) {
      int n0 = nt << 4, c = n0 + ll;
      v8f a0, a1; wmma_mm2(kvh, 256, wvp, 256, 256, n0, a0, a1);
      float bb = vb[c];
#pragma unroll
      for (int i = 0; i < 8; ++i) {
        AH[16384 + (i + (lh << 3)) * 256 + c]      = (h16)(a0[i] + bb);
        AH[16384 + (16 + i + (lh << 3)) * 256 + c] = (h16)(a1[i] + bb);
      }
    }
    __syncthreads();

    // scores per head: S = Q_h (32x32) @ K_h^T (32x32) / sqrt(32) -> FB
    for (int t = wv; t < 16; t += 8) {
      int head = t >> 1, n0 = (t & 1) << 4;
      v8f a0, a1;
      wmma_mm2(AH + head * 32, 256, AH + 8192 + head * 1024, 32, 32, n0, a0, a1);
#pragma unroll
      for (int i = 0; i < 8; ++i) {
        FB[(head * 32 + i + (lh << 3)) * 32 + n0 + ll]      = a0[i] * 0.17677669529f;
        FB[(head * 32 + 16 + i + (lh << 3)) * 32 + n0 + ll] = a1[i] * 0.17677669529f;
      }
    }
    __syncthreads();

    // softmax per row (256 rows of 32) -> f16 weights at AH[24576:32768)
    {
      float* row = FB + tid * 32;
      float m = -1e30f;
#pragma unroll
      for (int k = 0; k < 32; ++k) m = fmaxf(m, row[k]);
      float s = 0.f;
#pragma unroll
      for (int k = 0; k < 32; ++k) { float e = expf(row[k] - m); row[k] = e; s += e; }
      float inv = 1.f / s;
#pragma unroll
      for (int k = 0; k < 32; ++k) AH[24576 + tid * 32 + k] = (h16)(row[k] * inv);
    }
    __syncthreads();

    // out_h = w_h @ V_h -> attn-out f16 at AH[0:8192)
    for (int t = wv; t < 16; t += 8) {
      int head = t >> 1, n0 = (t & 1) << 4;
      v8f a0, a1;
      wmma_mm2(AH + 24576 + head * 1024, 32, AH + 16384 + head * 32, 256, 32, n0, a0, a1);
#pragma unroll
      for (int i = 0; i < 8; ++i) {
        AH[(i + (lh << 3)) * 256 + head * 32 + n0 + ll]      = (h16)a0[i];
        AH[(16 + i + (lh << 3)) * 256 + head * 32 + n0 + ll] = (h16)a1[i];
      }
    }
    __syncthreads();

    // O projection -> FB f32
    for (int nt = wv; nt < 16; nt += 8) {
      int n0 = nt << 4, c = n0 + ll;
      v8f a0, a1; wmma_mm2(AH, 256, wo, 256, 256, n0, a0, a1);
      float bb = ob[c];
#pragma unroll
      for (int i = 0; i < 8; ++i) {
        FB[(i + (lh << 3)) * 256 + c]      = a0[i] + bb;
        FB[(16 + i + (lh << 3)) * 256 + c] = a1[i] + bb;
      }
    }
    __syncthreads();
    // LN(out + q_in) -> RES (assign on d=0, accumulate on d=1)
    ln_rows(FB, qh, p.a_g + d * 256, p.a_beta + d * 256, nullptr, RES, d == 1);
    __syncthreads();
  }

  // ---- 6 Mamba blocks ----
  for (int ib = 0; ib < 6; ++ib) {
    const h16* wbase = ws + OFF_BLK + (size_t)ib * BLK_STRIDE;
    const h16 *w_in = wbase + B_IN, *w_gate = wbase + B_GATE, *w_sout = wbase + B_SOUT,
              *w_f1 = wbase + B_F1, *w_f2 = wbase + B_F2, *w_bm = wbase + B_BM,
              *w_ct = wbase + B_CT;

    // LN1 -> xln f16 BH[0:8192)
    ln_rows(RES, nullptr, p.m_ln1_g + ib * 256, p.m_ln1_b + ib * 256, BH, nullptr, false);
    __syncthreads();

    // gate = sigmoid(xln@Wg + b) -> FB
    for (int nt = wv; nt < 16; nt += 8) {
      int n0 = nt << 4, c = n0 + ll;
      v8f a0, a1; wmma_mm2(BH, 256, w_gate, 256, 256, n0, a0, a1);
      float bb = p.m_gate_b[ib * 256 + c];
#pragma unroll
      for (int i = 0; i < 8; ++i) {
        FB[(i + (lh << 3)) * 256 + c]      = 1.f / (1.f + expf(-(a0[i] + bb)));
        FB[(16 + i + (lh << 3)) * 256 + c] = 1.f / (1.f + expf(-(a1[i] + bb)));
      }
    }
    __syncthreads();

    // x_in = (xln@Win_upper + b) * gate -> f16 BH[8192:16384)
    for (int nt = wv; nt < 16; nt += 8) {
      int n0 = nt << 4, c = n0 + ll;
      v8f a0, a1; wmma_mm2(BH, 256, w_in, 256, 256, n0, a0, a1);
      float bb = p.m_in_b[ib * 512 + 256 + c];
#pragma unroll
      for (int i = 0; i < 8; ++i) {
        int r0 = i + (lh << 3), r1 = 16 + r0;
        BH[8192 + r0 * 256 + c] = (h16)((a0[i] + bb) * FB[r0 * 256 + c]);
        BH[8192 + r1 * 256 + c] = (h16)((a1[i] + bb) * FB[r1 * 256 + c]);
      }
    }
    __syncthreads();

    // xB = x_in @ Bm^T  (32x16) -> FB[0:512)
    for (int nt = wv; nt < 1; nt += 8) {
      v8f a0, a1; wmma_mm2(BH + 8192, 256, w_bm, 16, 256, 0, a0, a1);
#pragma unroll
      for (int i = 0; i < 8; ++i) {
        FB[(i + (lh << 3)) * 16 + ll]      = a0[i];
        FB[(16 + i + (lh << 3)) * 16 + ll] = a1[i];
      }
    }
    // zero padded state history Hs (32x32 f16) in AH[0:1024)
    for (int j = tid; j < 1024; j += 256) AH[j] = (h16)0.f;
    if (tid < 16) hbuf[tid] = 0.f;
    __syncthreads();

    // sequential SSM scan over 32 tokens, 16-state (f32)
    float arow[16];
    if (tid < 16) {
#pragma unroll
      for (int a = 0; a < 16; ++a) arow[a] = p.m_A[ib * 256 + tid * 16 + a];
    }
    for (int t = 0; t < 32; ++t) {
      float nh = 0.f;
      if (tid < 16) {
        nh = FB[t * 16 + tid];
#pragma unroll
        for (int a = 0; a < 16; ++a) nh += hbuf[a] * arow[a];
      }
      __syncthreads();
      if (tid < 16) { hbuf[tid] = nh; AH[t * 32 + tid] = (h16)nh; }
      __syncthreads();
    }

    // ys = Hs(32x32,padded) @ C^T(32x256,padded) -> f16 BH[0:8192)
    for (int nt = wv; nt < 16; nt += 8) {
      int n0 = nt << 4, c = n0 + ll;
      v8f a0, a1; wmma_mm2(AH, 32, w_ct, 256, 32, n0, a0, a1);
#pragma unroll
      for (int i = 0; i < 8; ++i) {
        BH[(i + (lh << 3)) * 256 + c]      = (h16)a0[i];
        BH[(16 + i + (lh << 3)) * 256 + c] = (h16)a1[i];
      }
    }
    __syncthreads();

    // RES += ys @ Wsout + b
    for (int nt = wv; nt < 16; nt += 8) {
      int n0 = nt << 4, c = n0 + ll;
      v8f a0, a1; wmma_mm2(BH, 256, w_sout, 256, 256, n0, a0, a1);
      float bb = p.m_sout_b[ib * 256 + c];
#pragma unroll
      for (int i = 0; i < 8; ++i) {
        RES[(i + (lh << 3)) * 256 + c]      += a0[i] + bb;
        RES[(16 + i + (lh << 3)) * 256 + c] += a1[i] + bb;
      }
    }
    __syncthreads();

    // LN2 -> f16 BH[0:8192)
    ln_rows(RES, nullptr, p.m_ln2_g + ib * 256, p.m_ln2_b + ib * 256, BH, nullptr, false);
    __syncthreads();

    // f1: (32x256)@(256x1024), gelu -> f16 AH[0:32768)
    for (int nt = wv; nt < 64; nt += 8) {
      int n0 = nt << 4, c = n0 + ll;
      v8f a0, a1; wmma_mm2(BH, 256, w_f1, 1024, 256, n0, a0, a1);
      float bb = p.m_f1_b[ib * 1024 + c];
#pragma unroll
      for (int i = 0; i < 8; ++i) {
        float v0 = a0[i] + bb, v1 = a1[i] + bb;
        AH[(i + (lh << 3)) * 1024 + c]      = (h16)gelu_exact(v0);
        AH[(16 + i + (lh << 3)) * 1024 + c] = (h16)gelu_exact(v1);
      }
    }
    __syncthreads();

    // f2: RES += (32x1024)@(1024x256)
    for (int nt = wv; nt < 16; nt += 8) {
      int n0 = nt << 4, c = n0 + ll;
      v8f a0, a1; wmma_mm2(AH, 1024, w_f2, 256, 1024, n0, a0, a1);
      float bb = p.m_f2_b[ib * 256 + c];
#pragma unroll
      for (int i = 0; i < 8; ++i) {
        RES[(i + (lh << 3)) * 256 + c]      += a0[i] + bb;
        RES[(16 + i + (lh << 3)) * 256 + c] += a1[i] + bb;
      }
    }
    __syncthreads();
  }

  // ---- final head: token mean -> LN(256) -> gelu(linear) ----
  float gv = 0.f;
  for (int t = 0; t < 32; ++t) gv += RES[t * 256 + tid];
  gv *= (1.f / 32.f);
  FB[256 + tid] = gv; __syncthreads();
  for (int s = 128; s; s >>= 1) { if (tid < s) FB[256 + tid] += FB[256 + tid + s]; __syncthreads(); }
  float mu = FB[256] * (1.f / 256.f); __syncthreads();
  float dv = gv - mu;
  FB[256 + tid] = dv * dv; __syncthreads();
  for (int s = 128; s; s >>= 1) { if (tid < s) FB[256 + tid] += FB[256 + tid + s]; __syncthreads(); }
  float var = FB[256] * (1.f / 256.f); __syncthreads();
  float xn = (gv - mu) * rsqrtf(var + 1e-5f) * p.fin_g[tid] + p.fin_beta[tid];
  FB[tid] = xn; __syncthreads();
  float oc = p.fin_b[tid];
  for (int k = 0; k < 256; ++k) oc += FB[k] * p.fin_w[tid * 256 + k];
  out[(size_t)bid * 256 + tid] = gelu_exact(oc);
  if (tid == 0) {
    // softmax rows each sum to 1 -> mean over (H,Lq,Lk) is exactly 1/Lk = 1/32 for both maps
    float wm = 1.f / 32.f;
    float ratio = wm / (wm + wm + 1e-6f);
    out[1024 * 256 + bid] = ratio;
    out[1024 * 256 + 1024 + bid] = ratio;
  }
}

// ---------------- host launcher ----------------
extern "C" void kernel_launch(void* const* d_in, const int* in_sizes, int n_in,
                              void* d_out, int out_size, void* d_ws, size_t ws_size,
                              hipStream_t stream) {
  h16* wsh = (h16*)d_ws;
  auto F = [&](int i) { return (const float*)d_in[i]; };

  auto CV = [&](const float* src, size_t off, int OUT, int K, int K2) {
    int n = K2 * OUT;
    convT<<<(n + 255) / 256, 256, 0, stream>>>(src, wsh + off, OUT, K, K2);
  };

  // projections
  CV(F(2), OFF_BP, 256, 128, 128);                 // bp_w
  CV(F(7), OFF_VP, 256, 256, 256);                 // vp_w
  // attention (2 dirs x q,k,v,o)
  for (int d = 0; d < 2; ++d) {
    CV(F(12) + (size_t)d * 65536, OFF_ATTN + (size_t)(d * 4 + 0) * 65536, 256, 256, 256);
    CV(F(14) + (size_t)d * 65536, OFF_ATTN + (size_t)(d * 4 + 1) * 65536, 256, 256, 256);
    CV(F(16) + (size_t)d * 65536, OFF_ATTN + (size_t)(d * 4 + 2) * 65536, 256, 256, 256);
    CV(F(18) + (size_t)d * 65536, OFF_ATTN + (size_t)(d * 4 + 3) * 65536, 256, 256, 256);
  }
  CV(F(41), OFF_FIN, 256, 256, 256);               // fin_w
  // mamba blocks
  for (int i = 0; i < 6; ++i) {
    size_t base = OFF_BLK + (size_t)i * BLK_STRIDE;
    CV(F(24) + (size_t)i * 131072 + 65536, base + B_IN,   256, 256, 256);   // m_in_w rows [256:512)
    CV(F(29) + (size_t)i * 65536,          base + B_GATE, 256, 256, 256);   // m_gate_w
    CV(F(31) + (size_t)i * 65536,          base + B_SOUT, 256, 256, 256);   // m_sout_w
    CV(F(35) + (size_t)i * 262144,         base + B_F1,  1024, 256, 256);   // m_f1_w
    CV(F(37) + (size_t)i * 262144,         base + B_F2,   256, 1024, 1024); // m_f2_w
    CV(F(27) + (size_t)i * 4096,           base + B_BM,    16, 256, 256);   // m_B
    CV(F(28) + (size_t)i * 4096,           base + B_CT,   256, 16, 32);     // m_C (K padded to 32)
  }

  Params p;
  p.ble = F(0); p.vis = F(1);
  p.bp_b = F(3); p.bp_emb = F(4); p.bp_g = F(5); p.bp_beta = F(6);
  p.vp_b = F(8); p.vp_emb = F(9); p.vp_g = F(10); p.vp_beta = F(11);
  p.a_qb = F(13); p.a_kb = F(15); p.a_vb = F(17); p.a_ob = F(19);
  p.a_g = F(20); p.a_beta = F(21);
  p.m_ln1_g = F(22); p.m_ln1_b = F(23); p.m_in_b = F(25); p.m_A = F(26);
  p.m_gate_b = F(30); p.m_sout_b = F(32); p.m_ln2_g = F(33); p.m_ln2_b = F(34);
  p.m_f1_b = F(36); p.m_f2_b = F(38);
  p.fin_g = F(39); p.fin_beta = F(40); p.fin_w = F(41); p.fin_b = F(42);

  fused_fwd<<<1024, 256, 0, stream>>>(p, (const h16*)d_ws, (float*)d_out);
}
